// MultiPerspectiveLayer_32976758899274
// MI455X (gfx1250) — matrix operand order repaired and checked by reading, not compiled, not run on hardware
//
#include <hip/hip_runtime.h>
#include <hip/hip_bf16.h>
#include <math.h>

// Problem constants (from the reference)
#define B_  32
#define TL_ 256
#define TR_ 256
#define D_  256
#define L_  32

// Padded LDS row stride for staged rt tiles: 256 halves + 8 halves (16B) pad
// = 264 halves = 528 B = 132 dwords -> bank starts 4*n mod 64, conflict-free.
#define SRT_STRIDE 264
#define CHUNK_ROWS 64   // rows of rt staged per LDS chunk (33,792 B)

typedef __attribute__((ext_vector_type(16))) _Float16 v16h;
typedef __attribute__((ext_vector_type(8)))  _Float16 v8h;
typedef __attribute__((ext_vector_type(8)))  float    v8f;

#if defined(__has_builtin)
#if __has_builtin(__builtin_amdgcn_tensor_load_to_lds) && \
    __has_builtin(__builtin_amdgcn_s_wait_tensorcnt)
#define USE_TDM 1
#endif
#endif
#ifndef USE_TDM
#define USE_TDM 0
#endif

// ---------------------------------------------------------------------------
// WMMA helpers (CDNA5 gfx1250, wave32).
// A fragment (16x32, f16, row-major src): lane l<16 -> row l, K = {k0..k0+7} in
// v0..3 and {k0+16..k0+23} in v4..7; lanes 16..31 same rows, K shifted by +8.
// B fragment (32x16, f16, B = srcT of a row-major matrix): lane n<16 -> column
// n (= src row n), same K layout as the A fragment (lanes 16..31: K +8).
// C/D: VGPR r, lane n -> element (M = r + 8*(n>=16), N = n&15).
// ---------------------------------------------------------------------------

static __device__ __forceinline__ v8f wmma_f16(v16h a, v16h b, v8f c) {
  return __builtin_amdgcn_wmma_f32_16x16x32_f16(false, a, false, b, (short)0, c,
                                                false, false);
}

static __device__ __forceinline__ v16h load_afrag(const _Float16* base, int ld) {
  const int lane = threadIdx.x & 31;
  const _Float16* src = base + (lane & 15) * ld + ((lane & 16) ? 8 : 0);
  v8h lo = *(const v8h*)(src);
  v8h hi = *(const v8h*)(src + 16);
  v16h f;
#pragma unroll
  for (int i = 0; i < 8; ++i) { f[i] = lo[i]; f[i + 8] = hi[i]; }
  return f;
}

static __device__ __forceinline__ v16h load_abcast(const _Float16* base) {
  const int lane = threadIdx.x & 31;
  const _Float16* src = base + ((lane & 16) ? 8 : 0);
  v8h lo = *(const v8h*)(src);
  v8h hi = *(const v8h*)(src + 16);
  v16h f;
#pragma unroll
  for (int i = 0; i < 8; ++i) { f[i] = lo[i]; f[i + 8] = hi[i]; }
  return f;
}

static __device__ __forceinline__ v16h load_bfrag(const _Float16* base, int ld) {
  const int lane = threadIdx.x & 31;
  const _Float16* src = base + (lane & 15) * ld + ((lane & 16) ? 8 : 0);
  v8h lo = *(const v8h*)(src);
  v8h hi = *(const v8h*)(src + 16);
  v16h f;
#pragma unroll
  for (int i = 0; i < 8; ++i) { f[i] = lo[i]; f[i + 8] = hi[i]; }
  return f;
}

// B fragment from the padded LDS staging buffer (rows relative to chunk).
static __device__ __forceinline__ v16h load_bfrag_lds(const _Float16* srt,
                                                      int row0, int k0) {
  const int lane = threadIdx.x & 31;
  const _Float16* src =
      srt + (row0 + (lane & 15)) * SRT_STRIDE + k0 + ((lane & 16) ? 8 : 0);
  v8h lo = *(const v8h*)(src);
  v8h hi = *(const v8h*)(src + 16);
  v16h f;
#pragma unroll
  for (int i = 0; i < 8; ++i) { f[i] = lo[i]; f[i + 8] = hi[i]; }
  return f;
}

#if USE_TDM
typedef unsigned int u32_;
typedef __attribute__((ext_vector_type(4))) u32_ u32x4_;
typedef __attribute__((ext_vector_type(8))) int  i32x8_;
typedef __attribute__((ext_vector_type(4))) int  i32x4_;

// Issue one TDM 2D tile load: CHUNK_ROWS rows x 512B from global (row-major,
// contiguous) into LDS, inserting a 16B pad after every 512B row
// (pad_interval = 128 DWORDs -> code 6, pad_amount = 4 DWORDs -> code 3).
// data_size = 8B units: row = 64 units. 6-arg builtin (clang-23 therock HIP):
// (u32x4 g0, i32x8 g1, i32x4 g2, i32x4 g3, i32x8 g4, i32 cpol).
static __device__ __forceinline__ void tdm_load_chunk(const _Float16* gsrc,
                                                      _Float16* lds_dst) {
  const unsigned long long ga = (unsigned long long)(uintptr_t)gsrc;
  const u32_ lds_addr = (u32_)(uintptr_t)lds_dst;  // low 32b = LDS offset

  u32x4_ g0;
  g0[0] = 1u;                                  // count=1 valid descriptor
  g0[1] = lds_addr;                            // D#.lds_addr (bytes)
  g0[2] = (u32_)(ga & 0xFFFFFFFFu);            // global_addr[31:0]
  g0[3] = (u32_)((ga >> 32) & 0x01FFFFFFu)     // global_addr[56:32]
        | (2u << 30);                          // type = 2 ("image")

  i32x8_ g1;
  g1[0] = (int)((3u << 16)                     // data_size = 8B
              | (1u << 20)                     // pad_enable
              | (6u << 22)                     // pad_interval: 128 DWORDs
              | (3u << 25));                   // pad_amount: 4 DWORDs
  g1[1] = (int)(64u << 16);                    // tensor_dim0 = 64 (lo16)
  g1[2] = (int)(((u32_)CHUNK_ROWS) << 16);     // tensor_dim1 = 64 (lo16)
  g1[3] = (int)(64u << 16);                    // tile_dim0 = 64
  g1[4] = (int)CHUNK_ROWS;                     // tile_dim1 = 64, tile_dim2 = 0
  g1[5] = 64;                                  // tensor_dim0_stride = 64 units
  g1[6] = 0;
  g1[7] = 0;

  i32x4_ gz4;
  gz4[0] = 0; gz4[1] = 0; gz4[2] = 0; gz4[3] = 0;  // dims 2..4 unused
  i32x8_ gz8;
#pragma unroll
  for (int i = 0; i < 8; ++i) gz8[i] = 0;

  __builtin_amdgcn_tensor_load_to_lds(g0, g1, gz4, gz4, gz8, 0);
  __builtin_amdgcn_s_wait_tensorcnt((unsigned short)0);
}
#endif

// ---------------------------------------------------------------------------
// Kernel 1: f32 -> f16 conversion of lt/rt + eps-clamped inverse row norms.
// ---------------------------------------------------------------------------
__global__ void __launch_bounds__(256)
prep_rows_kernel(const float* __restrict__ lt, const float* __restrict__ rt,
                 _Float16* __restrict__ lt16, _Float16* __restrict__ rt16,
                 float* __restrict__ inv_nl, float* __restrict__ inv_nr) {
  const int t = blockIdx.x, b = blockIdx.y, side = blockIdx.z;
  const int d = threadIdx.x;
  const float* src = side ? rt : lt;
  _Float16* dst = side ? rt16 : lt16;
  float* invp = side ? inv_nr : inv_nl;

  const size_t idx = ((size_t)(b * 256 + t)) * 256 + d;
  const float x = src[idx];
  dst[idx] = (_Float16)x;

  float s = x * x;
#pragma unroll
  for (int off = 16; off >= 1; off >>= 1) s += __shfl_xor(s, off, 32);

  __shared__ float part[8];
  if ((threadIdx.x & 31) == 0) part[threadIdx.x >> 5] = s;
  __syncthreads();
  if (threadIdx.x == 0) {
    float tot = 0.f;
#pragma unroll
    for (int i = 0; i < 8; ++i) tot += part[i];
    invp[b * 256 + t] = rsqrtf(fmaxf(tot, 1e-6f));
  }
}

// ---------------------------------------------------------------------------
// Kernel 2: h_rt concat + the four W matrices -> f16.
// ---------------------------------------------------------------------------
__global__ void prep_small_kernel(const float* __restrict__ fh,
                                  const float* __restrict__ bh,
                                  const float* __restrict__ wf,
                                  const float* __restrict__ wm,
                                  const float* __restrict__ wa,
                                  const float* __restrict__ wx,
                                  _Float16* __restrict__ h16,
                                  _Float16* __restrict__ w16) {
  const int idx = blockIdx.x * blockDim.x + threadIdx.x;
  if (idx < B_ * D_) {
    const int b = idx >> 8, d = idx & 255;
    const float v = (d < 128) ? fh[b * 128 + d] : bh[b * 128 + (d - 128)];
    h16[idx] = (_Float16)v;
  } else {
    const int widx = idx - B_ * D_;
    if (widx < 4 * L_ * D_) {
      const int which = widx / (L_ * D_);
      const int off = widx % (L_ * D_);
      const float* src = (which == 0) ? wf : (which == 1) ? wm
                         : (which == 2) ? wa : wx;
      w16[widx] = (_Float16)src[off];
    }
  }
}

// ---------------------------------------------------------------------------
// Kernel 3: cosine score matrix via WMMA.
// ---------------------------------------------------------------------------
__global__ void __launch_bounds__(128)
scores_kernel(const _Float16* __restrict__ lt16, const _Float16* __restrict__ rt16,
              const float* __restrict__ inv_nl, const float* __restrict__ inv_nr,
              _Float16* __restrict__ s16) {
  const int it = blockIdx.x, b = blockIdx.y;
  const int w = threadIdx.x >> 5;
  const _Float16* ltb = lt16 + ((size_t)b * TL_ + it * 16) * D_;
  const _Float16* rtb = rt16 + (size_t)b * TR_ * D_;

  v16h a[8];
#pragma unroll
  for (int k = 0; k < 8; ++k) a[k] = load_afrag(ltb + k * 32, D_);

  const int lane = threadIdx.x & 31;
  const int half = (lane & 16) ? 8 : 0;
  const int n = lane & 15;

  for (int jc = 0; jc < 4; ++jc) {
    const int jt = w * 4 + jc;
    v8f c;
#pragma unroll
    for (int i = 0; i < 8; ++i) c[i] = 0.f;
#pragma unroll
    for (int k = 0; k < 8; ++k) {
      v16h bf = load_bfrag(rtb + jt * 16 * D_ + k * 32, D_);
      c = wmma_f16(a[k], bf, c);
    }
    const float ir = inv_nr[b * TR_ + jt * 16 + n];
#pragma unroll
    for (int r = 0; r < 8; ++r) {
      const int row = it * 16 + r + half;
      const float il = inv_nl[b * TL_ + row];
      s16[((size_t)(b * TL_ + row)) * TR_ + jt * 16 + n] =
          (_Float16)(c[r] * il * ir);
    }
  }
}

// ---------------------------------------------------------------------------
// Kernel 4: att[b,i,d] = sum_j scores[b,i,j] * rt[b,j,d]   (~1 GFLOP, VALU).
// ---------------------------------------------------------------------------
__global__ void __launch_bounds__(256)
att_kernel(const _Float16* __restrict__ s16, const float* __restrict__ rt,
           _Float16* __restrict__ att16) {
  const int i = blockIdx.x, b = blockIdx.y, d = threadIdx.x;
  __shared__ float sj[TR_];
  sj[d] = (float)s16[((size_t)(b * TL_ + i)) * TR_ + d];
  __syncthreads();
  float acc = 0.f;
  const float* rtb = rt + (size_t)b * TR_ * D_ + d;
#pragma unroll 4
  for (int j = 0; j < TR_; ++j) acc += sj[j] * rtb[(size_t)j * D_];
  att16[((size_t)(b * TL_ + i)) * D_ + d] = (_Float16)acc;
}

// ---------------------------------------------------------------------------
// Kernel 5: max-pooling matching (the 34 GFLOP hot loop), v2.
// grid (8, B) x 128 threads: 4 waves per block, wave w -> l = bx*4 + w.
// rt[b] is staged into LDS in 64-row chunks (TDM async-tensor load when
// available), shared by all 4 waves; each wave register-blocks 2 i-tiles so
// every B fragment read from LDS feeds two WMMAs. Running max kept in VGPRs
// for all 16 i-tiles (tanh(max) == max(tanh), tanh applied at the end).
// ---------------------------------------------------------------------------
__global__ void __launch_bounds__(128)
maxpool_kernel(const _Float16* __restrict__ lt16, const _Float16* __restrict__ rt16,
               const _Float16* __restrict__ w16, float* __restrict__ out) {
  const int lg = blockIdx.x, b = blockIdx.y;
  const int w = threadIdx.x >> 5;
  const int l = lg * 4 + w;
  const _Float16* wrow = w16 + (size_t)1 * (L_ * D_) + (size_t)l * D_;  // W_maxpool
  const _Float16* ltb = lt16 + (size_t)b * TL_ * D_;
  const _Float16* rtb = rt16 + (size_t)b * TR_ * D_;

  __shared__ _Float16 srt[CHUNK_ROWS * SRT_STRIDE];  // 33,792 B

  v8f rmax[16];
#pragma unroll
  for (int t = 0; t < 16; ++t)
#pragma unroll
    for (int i = 0; i < 8; ++i) rmax[t][i] = -3.0e38f;

  for (int chunk = 0; chunk < TR_ / CHUNK_ROWS; ++chunk) {
    __syncthreads();  // previous chunk fully consumed
#if USE_TDM
    if (threadIdx.x < 32) {
      tdm_load_chunk(rtb + (size_t)chunk * CHUNK_ROWS * D_, srt);
    }
#else
    {
      const _Float16* src = rtb + (size_t)chunk * CHUNK_ROWS * D_;
      for (int u = threadIdx.x; u < CHUNK_ROWS * 32; u += 128) {
        const int row = u >> 5, seg = u & 31;  // 32 x 16B units per row
        *(v8h*)&srt[row * SRT_STRIDE + seg * 8] =
            *(const v8h*)(src + row * D_ + seg * 8);
      }
    }
#endif
    __syncthreads();  // chunk visible to all waves

#pragma unroll
    for (int ip = 0; ip < 8; ++ip) {  // pairs of i-tiles, register-blocked
      v16h a0[8], a1[8];
#pragma unroll
      for (int k = 0; k < 8; ++k) {
        const v16h wf = load_abcast(wrow + k * 32);
        a0[k] = load_afrag(ltb + (ip * 32 + 0) * D_ + k * 32, D_) * wf;
        a1[k] = load_afrag(ltb + (ip * 32 + 16) * D_ + k * 32, D_) * wf;
      }
      for (int jt = 0; jt < CHUNK_ROWS / 16; ++jt) {
        v8f c0, c1;
#pragma unroll
        for (int i = 0; i < 8; ++i) { c0[i] = 0.f; c1[i] = 0.f; }
#pragma unroll
        for (int k = 0; k < 8; ++k) {
          const v16h bf = load_bfrag_lds(srt, jt * 16, k * 32);
          c0 = wmma_f16(a0[k], bf, c0);
          c1 = wmma_f16(a1[k], bf, c1);
        }
#pragma unroll
        for (int i = 0; i < 8; ++i) {
          rmax[ip * 2 + 0][i] = fmaxf(rmax[ip * 2 + 0][i], c0[i]);
          rmax[ip * 2 + 1][i] = fmaxf(rmax[ip * 2 + 1][i], c1[i]);
        }
      }
    }
  }

  // Reduce max over N (16 lanes per half-wave), tanh, store column 32+l.
  const int half = (threadIdx.x & 16) ? 8 : 0;
#pragma unroll
  for (int t = 0; t < 16; ++t) {
#pragma unroll
    for (int i = 0; i < 8; ++i) {
      float v = rmax[t][i];
      v = fmaxf(v, __shfl_xor(v, 8, 32));
      v = fmaxf(v, __shfl_xor(v, 4, 32));
      v = fmaxf(v, __shfl_xor(v, 2, 32));
      v = fmaxf(v, __shfl_xor(v, 1, 32));
      rmax[t][i] = v;
    }
    if ((threadIdx.x & 15) == 0) {
#pragma unroll
      for (int i = 0; i < 8; ++i) {
        const int row = t * 16 + i + half;
        out[((size_t)(b * TL_ + row)) * 128 + 32 + l] = tanhf(rmax[t][i]);
      }
    }
  }
}

// ---------------------------------------------------------------------------
// Kernel 6: full / attentive / max-attentive projections via WMMA + tanh.
// ---------------------------------------------------------------------------
__global__ void __launch_bounds__(192)
final_kernel(const _Float16* __restrict__ lt16, const _Float16* __restrict__ att16,
             const _Float16* __restrict__ h16, const _Float16* __restrict__ w16,
             float* __restrict__ out) {
  const int it = blockIdx.x, b = blockIdx.y;
  const int w = threadIdx.x >> 5;
  const int o = w >> 1;       // 0=full, 1=attentive, 2=max-attentive
  const int lt2 = w & 1;      // which 16-wide half of L
  const int wslot = (o == 0) ? 0 : (o == 1) ? 2 : 3;
  const _Float16* Wsel = w16 + (size_t)wslot * (L_ * D_);
  const _Float16* ltb = lt16 + ((size_t)b * TL_ + it * 16) * D_;
  const _Float16* mb = (o == 0) ? (h16 + (size_t)b * D_)
                                : (att16 + ((size_t)b * TL_ + it * 16) * D_);

  v8f c;
#pragma unroll
  for (int i = 0; i < 8; ++i) c[i] = 0.f;

#pragma unroll
  for (int k = 0; k < 8; ++k) {
    v16h la = load_afrag(ltb + k * 32, D_);
    v16h mf = (o == 0) ? load_abcast(mb + k * 32)      // wave-uniform branch
                       : load_afrag(mb + k * 32, D_);
    la = la * mf;
    v16h bf = load_bfrag(Wsel + lt2 * 16 * D_ + k * 32, D_);  // B = W^T
    c = wmma_f16(la, bf, c);
  }

  const int lane = threadIdx.x & 31;
  const int half = (lane & 16) ? 8 : 0;
  const int n = lane & 15;
  const int colbase = (o == 0) ? 0 : (o == 1) ? 64 : 96;
#pragma unroll
  for (int r = 0; r < 8; ++r) {
    const int row = it * 16 + r + half;
    out[((size_t)(b * TL_) + row) * 128 + colbase + lt2 * 16 + n] = tanhf(c[r]);
  }
}

// ---------------------------------------------------------------------------
// Host-side launch. Workspace ~16.9 MB; every call identical (capture-safe).
// ---------------------------------------------------------------------------
extern "C" void kernel_launch(void* const* d_in, const int* in_sizes, int n_in,
                              void* d_out, int out_size, void* d_ws, size_t ws_size,
                              hipStream_t stream) {
  (void)in_sizes; (void)n_in; (void)out_size; (void)ws_size;

  const float* lstm_lt = (const float*)d_in[0];
  const float* fwd_h_rt = (const float*)d_in[6];
  const float* bwd_h_rt = (const float*)d_in[8];
  const float* lstm_rt = (const float*)d_in[5];
  const float* W_full = (const float*)d_in[10];
  const float* W_maxp = (const float*)d_in[11];
  const float* W_att = (const float*)d_in[12];
  const float* W_maxatt = (const float*)d_in[13];
  float* out = (float*)d_out;

  char* p = (char*)d_ws;
  _Float16* lt16 = (_Float16*)p;  p += (size_t)B_ * TL_ * D_ * 2;   // 4 MB
  _Float16* rt16 = (_Float16*)p;  p += (size_t)B_ * TR_ * D_ * 2;   // 4 MB
  _Float16* att16 = (_Float16*)p; p += (size_t)B_ * TL_ * D_ * 2;   // 4 MB
  _Float16* s16 = (_Float16*)p;   p += (size_t)B_ * TL_ * TR_ * 2;  // 4 MB
  float* inv_nl = (float*)p;      p += (size_t)B_ * TL_ * 4;
  float* inv_nr = (float*)p;      p += (size_t)B_ * TR_ * 4;
  _Float16* h16 = (_Float16*)p;   p += (size_t)B_ * D_ * 2;
  _Float16* w16 = (_Float16*)p;   p += (size_t)4 * L_ * D_ * 2;

  prep_rows_kernel<<<dim3(TL_, B_, 2), 256, 0, stream>>>(
      lstm_lt, lstm_rt, lt16, rt16, inv_nl, inv_nr);

  {
    const int total = B_ * D_ + 4 * L_ * D_;
    prep_small_kernel<<<(total + 255) / 256, 256, 0, stream>>>(
        fwd_h_rt, bwd_h_rt, W_full, W_maxp, W_att, W_maxatt, h16, w16);
  }

  scores_kernel<<<dim3(TL_ / 16, B_), 128, 0, stream>>>(
      lt16, rt16, inv_nl, inv_nr, s16);

  att_kernel<<<dim3(TL_, B_), 256, 0, stream>>>(s16, lstm_rt, att16);

  maxpool_kernel<<<dim3(8, B_), 128, 0, stream>>>(lt16, rt16, w16, out);

  final_kernel<<<dim3(TL_ / 16, B_), 192, 0, stream>>>(
      lt16, att16, h16, w16, out);
}
